// Net_89687507075960
// MI455X (gfx1250) — compile-verified, matrix operand order checked by reading, not run on hardware
//
#include <hip/hip_runtime.h>
#include <hip/hip_bf16.h>
#include <math.h>

// ---------------------------------------------------------------------------
// CDNA5 (gfx1250) MoE-ResNet forward.
// All heavy math runs through v_wmma_f32_16x16x32_bf16 (wave32 WMMA):
//   - convs as implicit GEMM (M=N*OH*OW, K=Cin*KH*KW, N=Cout), BN+ReLU fused.
//     Cout is always a multiple of 64 -> unguarded b128 weight loads.
//   - gate / expert FCs as M=32-specialized batched GEMMs (grid.z = expert),
//     templated on N%128==0 so the hot path has zero load masks.
//   - ping-pong LDS double buffering, ONE barrier per K-step, register
//     prefetch of the next tile overlaps HBM latency with the WMMA stream.
// ---------------------------------------------------------------------------

typedef __attribute__((ext_vector_type(16))) __bf16 v16bf;
typedef __attribute__((ext_vector_type(8)))  float  v8f;

union FragU { unsigned u[8]; v16bf v; };

__device__ __forceinline__ unsigned short f2bf(float f) {
  unsigned u = __float_as_uint(f);
  u += 0x7FFFu + ((u >> 16) & 1u);          // round-to-nearest-even
  return (unsigned short)(u >> 16);
}

// pack 4 floats -> 4 bf16 -> one 8B LDS store (dst 8B-aligned by construction)
__device__ __forceinline__ void store4bf(unsigned short* dst,
                                         float a, float b, float c, float d) {
  uint2 v;
  v.x = ((unsigned)f2bf(b) << 16) | f2bf(a);
  v.y = ((unsigned)f2bf(d) << 16) | f2bf(c);
  *(uint2*)dst = v;
}

// x / d via magic: mul = 2^34/d + 1 (host-computed); valid for x < 2^34/d
__device__ __forceinline__ unsigned fdiv34(unsigned x, unsigned mul) {
  return __umulhi(x, mul) >> 2;
}

#define BM 64
#define BN 64
#define BK 32
#define LDP (BK + 8)   // LDS row pad (ushort units): 80B row stride (8B aligned)

// ---------------- implicit-GEMM conv + folded BN + optional ReLU -----------
// REQUIRES: Cout % 64 == 0 (true for all layers: 64/128/256/512).
template<int KH, int KW>
__global__ __launch_bounds__(128) void conv_bn_act_wmma(
    const float* __restrict__ in, const float* __restrict__ wgt,
    const float* __restrict__ scale, const float* __restrict__ shift,
    float* __restrict__ out,
    int Cin, int H, int W, int Cout,
    int OH, int OW, int OHW, unsigned mOW, unsigned mOHW,
    int stride, int pad, int relu)
{
  __shared__ unsigned short As[2][BM][LDP];
  __shared__ unsigned short Bs[2][BN][LDP];
  const int Mtot = 32 * OHW;
  const int Ktot = Cin * KH * KW;
  const int tid = threadIdx.x;
  const int m0 = blockIdx.x * BM;
  const int n0 = blockIdx.y * BN;
  const int rr = tid >> 3;          // 0..15 : row-in-pass
  const int kg = (tid & 7) * 4;     // 0..28 : k-group of 4

  // decompose this thread's 4 output rows once (magic div, K-loop invariant)
  int a_ibase[4], a_oh[4], a_ow[4];
#pragma unroll
  for (int p = 0; p < 4; ++p) {
    unsigned m = (unsigned)(m0 + p * 16 + rr);
    unsigned img = fdiv34(m, mOHW);
    unsigned rem = m - img * (unsigned)OHW;
    unsigned oh  = fdiv34(rem, mOW);
    unsigned ow  = rem - oh * (unsigned)OW;
    a_ibase[p] = (int)img * Cin;          // row base in channel units
    a_oh[p] = (int)oh * stride - pad;     // pre-offset by pad
    a_ow[p] = (int)ow * stride - pad;
  }

  const int wv = tid >> 5, lane = tid & 31;
  const int wr = wv >> 1, wc = wv & 1;       // 2x2 wave grid
  const int half = lane >> 4, lc = lane & 15;
  const bool vecB = ((Ktot & 3) == 0);       // wave-uniform

  float av[4][4], bv[4][4];

  auto gather = [&](int k0) {
    // A: im2col gather into registers (loads batch, waits deferred)
#pragma unroll
    for (int p = 0; p < 4; ++p) {
      int mvalid = (m0 + p * 16 + rr) < Mtot;
#pragma unroll
      for (int q = 0; q < 4; ++q) {
        int k = k0 + kg + q;
        float val = 0.f;
        if (mvalid && k < Ktot) {
          int kw = k % KW; int t = k / KW; int kh = t % KH; int ci = t / KH;
          int ih = a_oh[p] + kh;
          int iw = a_ow[p] + kw;
          if (ih >= 0 && ih < H && iw >= 0 && iw < W)
            val = in[(((long)a_ibase[p] + ci) * H + ih) * W + iw];
        }
        av[p][q] = val;
      }
    }
    // B: weights, contiguous in k, Cout%64==0 -> no row guard
    if (vecB && (k0 + BK <= Ktot)) {
#pragma unroll
      for (int p = 0; p < 4; ++p) {
        int n = n0 + p * 16 + rr;
        float4 w4 = *(const float4*)(wgt + (long)n * Ktot + k0 + kg);
        bv[p][0] = w4.x; bv[p][1] = w4.y; bv[p][2] = w4.z; bv[p][3] = w4.w;
      }
    } else {
#pragma unroll
      for (int p = 0; p < 4; ++p) {
        int n = n0 + p * 16 + rr;
#pragma unroll
        for (int q = 0; q < 4; ++q) {
          int k = k0 + kg + q;
          bv[p][q] = (k < Ktot) ? wgt[(long)n * Ktot + k] : 0.f;
        }
      }
    }
    if (k0 + BK < Ktot)  // prefetch next weight K-slice (global_prefetch_b8)
      __builtin_prefetch(wgt + (long)(n0 + rr) * Ktot + k0 + BK + kg, 0, 1);
  };
  auto lds_store = [&](int buf) {
#pragma unroll
    for (int p = 0; p < 4; ++p) {
      int ri = p * 16 + rr;
      store4bf(&As[buf][ri][kg], av[p][0], av[p][1], av[p][2], av[p][3]);
      store4bf(&Bs[buf][ri][kg], bv[p][0], bv[p][1], bv[p][2], bv[p][3]);
    }
  };

  v8f acc[2][2] = {};

  gather(0);
  lds_store(0);
  __syncthreads();

  const int nsteps = (Ktot + BK - 1) / BK;
  for (int s = 0; s < nsteps; ++s) {
    const int cur = s & 1;
    const bool more = (s + 1 < nsteps);
    if (more) gather((s + 1) * BK);   // global loads overlap compute below

    FragU a[2], b[2];
#pragma unroll
    for (int mt = 0; mt < 2; ++mt) {
      int row = wr * 32 + mt * 16 + lc;
#pragma unroll
      for (int vv = 0; vv < 8; ++vv) {
        int k = (vv >> 2) * 16 + half * 8 + (vv & 3) * 2;
        a[mt].u[vv] = *(const unsigned*)&As[cur][row][k];
      }
    }
#pragma unroll
    for (int nt = 0; nt < 2; ++nt) {
      int col = wc * 32 + nt * 16 + lc;
#pragma unroll
      for (int vv = 0; vv < 8; ++vv) {
        int k = (vv >> 2) * 16 + half * 8 + (vv & 3) * 2;
        b[nt].u[vv] = *(const unsigned*)&Bs[cur][col][k];
      }
    }
#pragma unroll
    for (int mt = 0; mt < 2; ++mt)
#pragma unroll
      for (int nt = 0; nt < 2; ++nt)
        acc[mt][nt] = __builtin_amdgcn_wmma_f32_16x16x32_bf16(
            false, a[mt].v, false, b[nt].v, (short)0, acc[mt][nt], false, false);

    if (more) {
      lds_store(cur ^ 1);   // safe: everyone finished reading cur^1 last iter
      __syncthreads();      // single barrier per K-step
    }
  }

  // epilogue: BN scale/shift + relu + NCHW scatter (magic-div decomposition)
#pragma unroll
  for (int mt = 0; mt < 2; ++mt)
#pragma unroll
    for (int nt = 0; nt < 2; ++nt) {
      int nn = n0 + wc * 32 + nt * 16 + lc;      // always < Cout
#pragma unroll
      for (int r = 0; r < 8; ++r) {
        int mm = m0 + wr * 32 + mt * 16 + half * 8 + r;
        if (mm < Mtot) {
          float x = acc[mt][nt][r];
          x = x * scale[nn] + shift[nn];
          if (relu) x = fmaxf(x, 0.f);
          unsigned img = fdiv34((unsigned)mm, mOHW);
          unsigned rem = (unsigned)mm - img * (unsigned)OHW;
          out[((long)img * Cout + nn) * OHW + rem] = x;
        }
      }
    }
}

// ---------------- M=32 batched GEMM (+bias) for gate & experts -------------
// out[e][m][n] = sum_k A[e][m][k] * Bw[e][n][k] + bias[e][n]   (torch [out,in])
// REQUIRES: K % 32 == 0 (all callers: K in {256, 512}).
// NFULL: N % 128 == 0 -> no bounds masks anywhere.
#define GBN 128
template<bool NFULL>
__global__ __launch_bounds__(128) void gemm_bias_wmma_m32(
    const float* __restrict__ A, const float* __restrict__ Bw,
    const float* __restrict__ bias, float* __restrict__ out,
    int N, int K, long aStride, long bStride, long biasStride, long outStride)
{
  __shared__ unsigned short As[2][32][LDP];
  __shared__ unsigned short Bs[2][GBN][LDP];
  const int e = blockIdx.z;
  const float* Ae = A + (long)e * aStride;
  const float* Be = Bw + (long)e * bStride;
  const int tid = threadIdx.x;
  const int n0 = blockIdx.y * GBN;
  const int rr = tid >> 3, kg = (tid & 7) * 4;
  const int wv = tid >> 5, lane = tid & 31;
  const int half = lane >> 4, lc = lane & 15;

  float4 a4[2], b4[8];
  auto gather = [&](int k0) {
#pragma unroll
    for (int p = 0; p < 2; ++p)              // A: 32 x 32, always in-bounds
      a4[p] = *(const float4*)(Ae + (long)(p * 16 + rr) * K + k0 + kg);
#pragma unroll
    for (int p = 0; p < 8; ++p) {            // B: 128 x 32
      int n = n0 + p * 16 + rr;
      if (NFULL) {
        b4[p] = *(const float4*)(Be + (long)n * K + k0 + kg);
      } else {
        b4[p] = make_float4(0.f, 0.f, 0.f, 0.f);
        if (n < N) b4[p] = *(const float4*)(Be + (long)n * K + k0 + kg);
      }
    }
    if (k0 + BK < K)
      __builtin_prefetch(Be + (long)(n0 + rr) * K + k0 + BK + kg, 0, 1);
  };
  auto lds_store = [&](int buf) {
#pragma unroll
    for (int p = 0; p < 2; ++p)
      store4bf(&As[buf][p * 16 + rr][kg], a4[p].x, a4[p].y, a4[p].z, a4[p].w);
#pragma unroll
    for (int p = 0; p < 8; ++p)
      store4bf(&Bs[buf][p * 16 + rr][kg], b4[p].x, b4[p].y, b4[p].z, b4[p].w);
  };

  v8f acc[2][2] = {};

  gather(0);
  lds_store(0);
  __syncthreads();

  const int nsteps = K / BK;
  for (int s = 0; s < nsteps; ++s) {
    const int cur = s & 1;
    const bool more = (s + 1 < nsteps);
    if (more) gather((s + 1) * BK);

    FragU a[2], b[2];
#pragma unroll
    for (int mt = 0; mt < 2; ++mt) {
      int row = mt * 16 + lc;
#pragma unroll
      for (int vv = 0; vv < 8; ++vv) {
        int k = (vv >> 2) * 16 + half * 8 + (vv & 3) * 2;
        a[mt].u[vv] = *(const unsigned*)&As[cur][row][k];
      }
    }
#pragma unroll
    for (int nt = 0; nt < 2; ++nt) {
      int col = wv * 32 + nt * 16 + lc;
#pragma unroll
      for (int vv = 0; vv < 8; ++vv) {
        int k = (vv >> 2) * 16 + half * 8 + (vv & 3) * 2;
        b[nt].u[vv] = *(const unsigned*)&Bs[cur][col][k];
      }
    }
#pragma unroll
    for (int mt = 0; mt < 2; ++mt)
#pragma unroll
      for (int nt = 0; nt < 2; ++nt)
        acc[mt][nt] = __builtin_amdgcn_wmma_f32_16x16x32_bf16(
            false, a[mt].v, false, b[nt].v, (short)0, acc[mt][nt], false, false);

    if (more) {
      lds_store(cur ^ 1);
      __syncthreads();
    }
  }

#pragma unroll
  for (int mt = 0; mt < 2; ++mt)
#pragma unroll
    for (int nt = 0; nt < 2; ++nt) {
      int nn = n0 + wv * 32 + nt * 16 + lc;
      if (NFULL || nn < N) {
#pragma unroll
        for (int r = 0; r < 8; ++r) {
          int mm = mt * 16 + half * 8 + r;     // M == 32, always in range
          float x = acc[mt][nt][r];
          if (bias) x += bias[(long)e * biasStride + nn];
          out[(long)e * outStride + (long)mm * N + nn] = x;
        }
      }
    }
}

// ---------------- small elementwise / reduction kernels --------------------
__global__ void bn_fold_kernel(const float* b, const float* g, const float* m,
                               const float* v, float* scale, float* shift, int C) {
  int c = blockIdx.x * blockDim.x + threadIdx.x;
  if (c >= C) return;
  float s = g[c] * rsqrtf(v[c] + 1e-5f);
  scale[c] = s; shift[c] = b[c] - m[c] * s;
}

// grid: (ceil(OH*OW/256), C, N) -> no divisions except one magic oh/ow split
__global__ void maxpool_kernel(const float* in, float* out, int H, int W,
                               int OH, int OW, unsigned mOW) {
  int t = blockIdx.x * blockDim.x + threadIdx.x;
  int c = blockIdx.y, n = blockIdx.z, C = gridDim.y;
  if (t >= OH * OW) return;
  unsigned oh = fdiv34((unsigned)t, mOW);
  unsigned ow = (unsigned)t - oh * (unsigned)OW;
  const float* plane = in + ((long)n * C + c) * H * W;
  float mx = -3.4e38f;
  for (int kh = 0; kh < 3; ++kh) {
    int ih = (int)oh * 2 - 1 + kh; if (ih < 0 || ih >= H) continue;
    for (int kw = 0; kw < 3; ++kw) {
      int iw = (int)ow * 2 - 1 + kw; if (iw < 0 || iw >= W) continue;
      mx = fmaxf(mx, plane[(long)ih * W + iw]);
    }
  }
  out[((long)n * C + c) * OH * OW + t] = mx;
}

__global__ __launch_bounds__(128) void avgpool_kernel(const float* in, float* out, int HW) {
  __shared__ float sh[128];
  long base = (long)blockIdx.x * HW;
  float s = 0.f;
  for (int i = threadIdx.x; i < HW; i += 128) s += in[base + i];
  sh[threadIdx.x] = s; __syncthreads();
  for (int o = 64; o > 0; o >>= 1) {
    if ((int)threadIdx.x < o) sh[threadIdx.x] += sh[threadIdx.x + o];
    __syncthreads();
  }
  if (threadIdx.x == 0) out[blockIdx.x] = sh[0] / (float)HW;
}

__global__ void fc_small_kernel(const float* in, const float* w, float* out,
                                int I, int O, int act) {
  int bb = blockIdx.x;
  for (int o = threadIdx.x; o < O; o += blockDim.x) {
    float s = 0.f;
    for (int i = 0; i < I; ++i) s += in[(long)bb * I + i] * w[(long)o * I + i];
    if (act == 1) s = fmaxf(s, 0.f);
    else if (act == 2) s = 1.f / (1.f + expf(-s));
    out[(long)bb * O + o] = s;
  }
}

// grid: (ceil(HW/256), C, B) -> zero index divisions
__global__ void se_apply_kernel(const float* x, const float* gate, const float* sc,
                                float* out, int HW) {
  int h = blockIdx.x * blockDim.x + threadIdx.x;
  int c = blockIdx.y, b = blockIdx.z, C = gridDim.y;
  if (h >= HW) return;
  long idx = ((long)b * C + c) * HW + h;
  float v = x[idx] * gate[(long)b * C + c] + sc[idx];
  out[idx] = fmaxf(v, 0.f);
}

__global__ __launch_bounds__(256) void ln_act_kernel(float* x, const float* g,
                                                     const float* b, int cols,
                                                     int rows_per_group, int do_gelu) {
  __shared__ float s1[256], s2[256];
  int row = blockIdx.x;
  float* xr = x + (long)row * cols;
  float sum = 0.f, sq = 0.f;
  for (int c = threadIdx.x; c < cols; c += 256) { float v = xr[c]; sum += v; sq += v * v; }
  s1[threadIdx.x] = sum; s2[threadIdx.x] = sq;
  __syncthreads();
  for (int o = 128; o > 0; o >>= 1) {
    if ((int)threadIdx.x < o) { s1[threadIdx.x] += s1[threadIdx.x + o]; s2[threadIdx.x] += s2[threadIdx.x + o]; }
    __syncthreads();
  }
  float mu = s1[0] / cols;
  float var = s2[0] / cols - mu * mu;
  float inv = rsqrtf(var + 1e-5f);
  long poff = rows_per_group > 0 ? (long)(row / rows_per_group) * cols : 0;
  for (int c = threadIdx.x; c < cols; c += 256) {
    float y = (xr[c] - mu) * inv * g[poff + c] + b[poff + c];
    if (do_gelu) y = 0.5f * y * (1.f + erff(y * 0.70710678118654752f));
    xr[c] = y;
  }
}

__global__ void logits_kernel(const float* h, const float* w, const float* bias,
                              const float* temp, float* out) {
  int idx = threadIdx.x; if (idx >= 256) return;
  int b = idx >> 3, e = idx & 7;
  float s = bias[e];
  for (int i = 0; i < 128; ++i) s += h[b * 128 + i] * w[e * 128 + i];
  float t = fmaxf(temp[0], 0.1f);
  out[b * 8 + e] = s / t;
}

__global__ void top2_kernel(const float* logits, float* gates) {
  int b = threadIdx.x; if (b >= 32) return;
  const float* l = logits + b * 8;
  int i1 = 0; float v1 = l[0];
  for (int e = 1; e < 8; ++e) if (l[e] > v1) { v1 = l[e]; i1 = e; }
  int i2 = (i1 == 0) ? 1 : 0; float v2 = l[i2];
  for (int e = 0; e < 8; ++e) if (e != i1 && l[e] > v2) { v2 = l[e]; i2 = e; }
  float e1 = 1.f, e2 = expf(v2 - v1);
  float s = e1 + e2;
  float g1 = e1 / s, g2 = e2 / s;
  float denom = g1 + g2 + 1e-8f;
  g1 /= denom; g2 /= denom;
  float* g = gates + b * 8;
  for (int e = 0; e < 8; ++e) g[e] = 0.f;
  g[i1] = g1; g[i2] = g2;
}

__global__ void combine_kernel(const float* eo, const float* gates, float* out) {
  int idx = blockIdx.x * blockDim.x + threadIdx.x;
  if (idx >= 32 * 1000) return;
  int b = idx / 1000, o = idx % 1000;
  float s = 0.f;
  for (int e = 0; e < 8; ++e)
    s += gates[b * 8 + e] * eo[((long)e * 32 + b) * 1000 + o];
  out[idx] = s;
}

// ---------------------------------------------------------------------------
// Host orchestration.
// Param flatten assumption: JAX pytree (dict keys sorted recursively) for the
// 'params' subtree: experts -> fe -> gate. 'x' detected by size (first or last).
// ---------------------------------------------------------------------------
extern "C" void kernel_launch(void* const* d_in, const int* in_sizes, int n_in,
                              void* d_out, int out_size, void* d_ws, size_t ws_size,
                              hipStream_t stream) {
  int xIdx, pBase;
  if (in_sizes[0] == 32 * 3 * 224 * 224) { xIdx = 0; pBase = 1; }
  else { xIdx = n_in - 1; pBase = 0; }

  auto IN = [&](int pi) { return (const float*)d_in[pBase + pi]; };
  auto magic34 = [](unsigned d) { return (unsigned)(17179869184ull / d + 1); };

  // ---- param indices (relative, sorted-key flatten of params) ----
  const int E_B1=0,E_B2=1,E_B3=2,E_B4=3,E_LN1B=4,E_LN1G=5,E_LN2B=6,E_LN2G=7,
            E_LN3B=8,E_LN3G=9,E_W1=10,E_W2=11,E_W3=12,E_W4=13;
  const int FE_BN1=14, FE_CONV1=18;
  // res blocks: {bn1, bn2, c1, c2, sc, sc_bn, se1, se2, Cout, stride}
  const int BLK[8][10] = {
    { 19, 23, 27, 28, -1, -1, 29, 30,  64, 1},
    { 31, 35, 39, 40, -1, -1, 41, 42,  64, 1},
    { 43, 47, 51, 52, 53, 54, 58, 59, 128, 2},
    { 60, 64, 68, 69, -1, -1, 70, 71, 128, 1},
    { 72, 76, 80, 81, 82, 83, 87, 88, 256, 2},
    { 89, 93, 97, 98, -1, -1, 99,100, 256, 1},
    {101,105,109,110,111,112,116,117, 512, 2},
    {118,122,126,127, -1, -1,128,129, 512, 1},
  };
  const int G_B1=130,G_B2=131,G_B3=132,G_LN1B=133,G_LN1G=134,G_LN2B=135,G_LN2G=136,
            G_TEMP=137,G_W1=138,G_W2=139,G_W3=140;

  // ---- workspace layout (floats) ----
  float* Wp = (float*)d_ws;
  float* bufA   = Wp;               // 25,700,000 (conv1 out + scratch)
  float* bufB   = Wp + 25700000;    // 6,500,000  (current activation)
  float* bufC   = Wp + 32200000;    // 6,500,000  (conv2 out)
  long S = 38700000;
  float* bn_scale = Wp + S;         S += 512;
  float* bn_shift = Wp + S;         S += 512;
  float* se_pool  = Wp + S;         S += 32 * 512;
  float* se_mid   = Wp + S;         S += 32 * 32;
  float* se_gate  = Wp + S;         S += 32 * 512;
  float* feats    = Wp + S;         S += 32 * 512;
  float* g_h1     = Wp + S;         S += 32 * 256;
  float* g_h2     = Wp + S;         S += 32 * 128;
  float* logits   = Wp + S;         S += 32 * 8;
  float* gates    = Wp + S;         S += 32 * 8;
  float* e_t1     = Wp + S;         S += 8 * 32 * 512;
  float* e_t2     = Wp + S;         S += 8 * 32 * 512;
  float* e_t3     = Wp + S;         S += 8 * 32 * 256;
  float* e_out    = Wp + S;         S += 8 * 32 * 1000;
  (void)ws_size; (void)out_size;

  auto fold = [&](int bnBase, int C) {
    bn_fold_kernel<<<(C + 255) / 256, 256, 0, stream>>>(
        IN(bnBase), IN(bnBase + 1), IN(bnBase + 2), IN(bnBase + 3),
        bn_scale, bn_shift, C);
  };
  auto conv = [&](const float* src, int Cin, int H, int Wd, int widx, int Cout,
                  int KHW, int stride, int pad, int bnBase, int relu,
                  float* dst, int OH, int OW) {
    fold(bnBase, Cout);
    int OHW = OH * OW;
    int M = 32 * OHW;
    unsigned mOW = magic34((unsigned)OW), mOHW = magic34((unsigned)OHW);
    dim3 g((M + BM - 1) / BM, (Cout + BN - 1) / BN);
    if (KHW == 7)
      conv_bn_act_wmma<7,7><<<g, 128, 0, stream>>>(src, IN(widx), bn_scale, bn_shift,
          dst, Cin, H, Wd, Cout, OH, OW, OHW, mOW, mOHW, stride, pad, relu);
    else if (KHW == 3)
      conv_bn_act_wmma<3,3><<<g, 128, 0, stream>>>(src, IN(widx), bn_scale, bn_shift,
          dst, Cin, H, Wd, Cout, OH, OW, OHW, mOW, mOHW, stride, pad, relu);
    else
      conv_bn_act_wmma<1,1><<<g, 128, 0, stream>>>(src, IN(widx), bn_scale, bn_shift,
          dst, Cin, H, Wd, Cout, OH, OW, OHW, mOW, mOHW, stride, pad, relu);
  };

  // ---- feature extractor ----
  const float* x = (const float*)d_in[xIdx];
  conv(x, 3, 224, 224, FE_CONV1, 64, 7, 2, 3, FE_BN1, 1, bufA, 112, 112);
  {
    dim3 g((56 * 56 + 255) / 256, 64, 32);
    maxpool_kernel<<<g, 256, 0, stream>>>(bufA, bufB, 112, 112, 56, 56, magic34(56u));
  }

  int C = 64, H = 56, Wd = 56;
  for (int blk = 0; blk < 8; ++blk) {
    int bn1 = BLK[blk][0], bn2 = BLK[blk][1], c1 = BLK[blk][2], c2 = BLK[blk][3];
    int sc = BLK[blk][4], sc_bn = BLK[blk][5], se1 = BLK[blk][6], se2 = BLK[blk][7];
    int Cout = BLK[blk][8], stride = BLK[blk][9];
    int OH = (H + 2 - 3) / stride + 1;
    int OW = (Wd + 2 - 3) / stride + 1;

    conv(bufB, C, H, Wd, c1, Cout, 3, stride, 1, bn1, 1, bufA, OH, OW);
    conv(bufA, Cout, OH, OW, c2, Cout, 3, 1, 1, bn2, 0, bufC, OH, OW);

    avgpool_kernel<<<32 * Cout, 128, 0, stream>>>(bufC, se_pool, OH * OW);
    fc_small_kernel<<<32, 128, 0, stream>>>(se_pool, IN(se1), se_mid, Cout, Cout / 16, 1);
    fc_small_kernel<<<32, 128, 0, stream>>>(se_mid, IN(se2), se_gate, Cout / 16, Cout, 2);

    const float* scp;
    if (sc >= 0) {
      conv(bufB, C, H, Wd, sc, Cout, 1, stride, 0, sc_bn, 0, bufA, OH, OW);
      scp = bufA;
    } else {
      scp = bufB;
    }
    dim3 g((OH * OW + 255) / 256, Cout, 32);
    se_apply_kernel<<<g, 256, 0, stream>>>(bufC, se_gate, scp, bufB, OH * OW);
    C = Cout; H = OH; Wd = OW;
  }
  avgpool_kernel<<<32 * 512, 128, 0, stream>>>(bufB, feats, H * Wd);  // [32,512]

  // ---- gate ----
  {
    dim3 g1(1, 2, 1);
    gemm_bias_wmma_m32<true><<<g1, 128, 0, stream>>>(feats, IN(G_W1), IN(G_B1), g_h1,
        256, 512, 0, 0, 0, 0);
    ln_act_kernel<<<32, 256, 0, stream>>>(g_h1, IN(G_LN1G), IN(G_LN1B), 256, 0, 1);
    dim3 g2(1, 1, 1);
    gemm_bias_wmma_m32<true><<<g2, 128, 0, stream>>>(g_h1, IN(G_W2), IN(G_B2), g_h2,
        128, 256, 0, 0, 0, 0);
    ln_act_kernel<<<32, 256, 0, stream>>>(g_h2, IN(G_LN2G), IN(G_LN2B), 128, 0, 1);
    logits_kernel<<<1, 256, 0, stream>>>(g_h2, IN(G_W3), IN(G_B3), IN(G_TEMP), logits);
    top2_kernel<<<1, 32, 0, stream>>>(logits, gates);
  }

  // ---- experts (batched over grid.z = 8) ----
  {
    dim3 ge(1, 4, 8);
    gemm_bias_wmma_m32<true><<<ge, 128, 0, stream>>>(feats, IN(E_W1), IN(E_B1), e_t1,
        512, 512, 0, 512L * 512, 512, 32L * 512);
    ln_act_kernel<<<8 * 32, 256, 0, stream>>>(e_t1, IN(E_LN1G), IN(E_LN1B), 512, 32, 1);

    gemm_bias_wmma_m32<true><<<ge, 128, 0, stream>>>(e_t1, IN(E_W2), IN(E_B2), e_t2,
        512, 512, 32L * 512, 512L * 512, 512, 32L * 512);
    ln_act_kernel<<<8 * 32, 256, 0, stream>>>(e_t2, IN(E_LN2G), IN(E_LN2B), 512, 32, 1);

    dim3 g3(1, 2, 8);
    gemm_bias_wmma_m32<true><<<g3, 128, 0, stream>>>(e_t2, IN(E_W3), IN(E_B3), e_t3,
        256, 512, 32L * 512, 256L * 512, 256, 32L * 256);
    ln_act_kernel<<<8 * 32, 256, 0, stream>>>(e_t3, IN(E_LN3G), IN(E_LN3B), 256, 32, 1);

    dim3 g4(1, 8, 8);
    gemm_bias_wmma_m32<false><<<g4, 128, 0, stream>>>(e_t3, IN(E_W4), IN(E_B4), e_out,
        1000, 256, 32L * 256, 1000L * 256, 1000, 32L * 1000);
  }

  // ---- gated combine ----
  combine_kernel<<<(32 * 1000 + 255) / 256, 256, 0, stream>>>(e_out, gates, (float*)d_out);
}
